// GPPT_75496935129278
// MI455X (gfx1250) — matrix operand order repaired and checked by reading, not compiled, not run on hardware
//
#include <hip/hip_runtime.h>
#include <hip/hip_bf16.h>

// ---------------------------------------------------------------------------
// Types for CDNA5 WMMA (wave32): v_wmma_f32_16x16x32_bf16
// ---------------------------------------------------------------------------
typedef __attribute__((ext_vector_type(16))) __bf16 v16bf;
typedef __attribute__((ext_vector_type(8)))  __bf16 v8bf;
typedef __attribute__((ext_vector_type(8)))  float  v8f;
typedef __attribute__((ext_vector_type(4)))  float  v4f;
typedef __attribute__((ext_vector_type(4)))  int    v4i;

// CDNA5 async global->LDS path (guarded: falls back to ds staging if absent)
#if __has_builtin(__builtin_amdgcn_global_load_async_to_lds_b128) && \
    __has_builtin(__builtin_amdgcn_s_wait_asynccnt)
#define GPPT_ASYNC_LDS 1
typedef __attribute__((address_space(1))) v4i* gppt_gp128_t;  // global v4i32*
typedef __attribute__((address_space(3))) v4i* gppt_lp128_t;  // LDS v4i32*
#else
#define GPPT_ASYNC_LDS 0
#endif

// ---------------------------------------------------------------------------
// fp32 -> bf16 conversion, 8 elements / thread. Non-temporal f32 reads so the
// 256MB fp32 adj doesn't evict the 128MB bf16 copy we want L2-resident.
// ---------------------------------------------------------------------------
__global__ __launch_bounds__(256)
void gppt_cvt_f32_bf16(const float* __restrict__ src, __bf16* __restrict__ dst,
                       long long n) {
  long long i = ((long long)blockIdx.x * blockDim.x + threadIdx.x) * 8;
  if (i + 7 < n) {
    v4f a = __builtin_nontemporal_load((const v4f*)(src + i));
    v4f b = __builtin_nontemporal_load((const v4f*)(src + i + 4));
    v8bf o;
    o[0] = (__bf16)a[0]; o[1] = (__bf16)a[1];
    o[2] = (__bf16)a[2]; o[3] = (__bf16)a[3];
    o[4] = (__bf16)b[0]; o[5] = (__bf16)b[1];
    o[6] = (__bf16)b[2]; o[7] = (__bf16)b[3];
    *(v8bf*)(dst + i) = o;
  }
}

// fp32 [K,N] -> bf16 TRANSPOSED [N,K] (for the tiny 512x512 weights)
__global__ __launch_bounds__(256)
void gppt_cvt_f32_bf16_T(const float* __restrict__ src, __bf16* __restrict__ dst,
                         int Ks, int Ns) {
  int idx = blockIdx.x * 256 + threadIdx.x;
  int k = idx / Ns, n = idx % Ns;
  dst[(size_t)n * Ks + k] = (__bf16)src[idx];
}

// ---------------------------------------------------------------------------
// Tiled bf16 WMMA GEMM:  C = act(A[M,K] @ Bt[N,K]^T + bias)
//   B operand consumed PRE-TRANSPOSED [N,K] so both tiles stage as contiguous
//   16B rows via async global->LDS (no in-kernel transpose).
//   Block tile 256x128, K-step 32; 256 threads = 8 waves (4x2 wave grid);
//   wave tile 64x64 = 4x4 WMMA tiles -> 16 v_wmma per K-step, 16 ds_load_b128
//   per 16 wmma. Double-buffered LDS (60KB of 320KB/WGP), 1 barrier/K-step.
//   LDS rows padded to 40 elems -> conflict-free b128 reads (bank 20*l % 64).
// ---------------------------------------------------------------------------
#define BM 256
#define BN 128
#define BK 32
#define LDT 40

__device__ __forceinline__
void gppt_copy16(const __bf16* gp, __bf16* lp) {
#if GPPT_ASYNC_LDS
  __builtin_amdgcn_global_load_async_to_lds_b128(
      (gppt_gp128_t)(gp), (gppt_lp128_t)(lp), 0, 0);
#else
  *(v8bf*)lp = *(const v8bf*)gp;
#endif
}

__device__ __forceinline__
void gppt_stage(const __bf16* __restrict__ A, const __bf16* __restrict__ Bt,
                __bf16* sA, __bf16* sB,
                int t, int blockM, int blockN, int K, int k0) {
  // ---- A tile: 256 rows x 32 cols, 16B per op, 4 ops/thread ----
#pragma unroll
  for (int i = 0; i < 4; ++i) {
    int q = t + i * 256;
    int row = q >> 2;            // 0..255
    int seg = q & 3;             // 0..3 (8 bf16 = 16B)
    gppt_copy16(A + (size_t)(blockM + row) * K + k0 + seg * 8,
                sA + row * LDT + seg * 8);
  }
  // ---- B tile: 128 rows (of Bt) x 32 cols, 2 ops/thread ----
#pragma unroll
  for (int i = 0; i < 2; ++i) {
    int q = t + i * 256;
    int row = q >> 2;            // 0..127
    int seg = q & 3;
    gppt_copy16(Bt + (size_t)(blockN + row) * K + k0 + seg * 8,
                sB + row * LDT + seg * 8);
  }
}

// WBMODE: 0 = no bf16 output, 1 = row-major [M,N], 2 = transposed [N,M]
template <bool HAS_BIAS, bool RELU, bool WF, int WBMODE>
__global__ __launch_bounds__(256)
void gppt_gemm_bf16(const __bf16* __restrict__ A,   // [M,K] row-major
                    const __bf16* __restrict__ Bt,  // [N,K] row-major (B^T)
                    const float*  __restrict__ bias,// [N]
                    float*   __restrict__ outF,     // [M,N] f32
                    __bf16*  __restrict__ outB,     // bf16 per WBMODE
                    int M, int N, int K)
{
  __shared__ __bf16 lsA[2][BM * LDT];   // 40 KB
  __shared__ __bf16 lsB[2][BN * LDT];   // 20 KB

  const int t     = threadIdx.x;
  const int lane  = t & 31;
  const int wave  = t >> 5;
  const int waveM = wave & 3;     // 4 waves along M, 64 rows each
  const int waveN = wave >> 2;    // 2 waves along N, 64 cols each
  const int half  = lane >> 4;    // 0/1: K-half selector per ISA operand layout
  const int l     = lane & 15;

  const int blockM = blockIdx.x * BM;
  const int blockN = blockIdx.y * BN;

  v8f acc[4][4];
#pragma unroll
  for (int mt = 0; mt < 4; ++mt)
#pragma unroll
    for (int nt = 0; nt < 4; ++nt)
      acc[mt][nt] = (v8f){0.f,0.f,0.f,0.f,0.f,0.f,0.f,0.f};

  const int nk = K / BK;
  gppt_stage(A, Bt, lsA[0], lsB[0], t, blockM, blockN, K, 0);

  for (int kk = 0; kk < nk; ++kk) {
    const int cur = kk & 1;
#if GPPT_ASYNC_LDS
    __builtin_amdgcn_s_wait_asynccnt(0);   // buf[cur] async stores landed
#endif
    __syncthreads();                       // (includes dscnt wait for ds stores)

    if (kk + 1 < nk)                       // uniform branch: EXEC stays full
      gppt_stage(A, Bt, lsA[cur ^ 1], lsB[cur ^ 1], t, blockM, blockN, K,
                 (kk + 1) * BK);

    // ---- fragments (ISA 16-bit operand lane layouts) ----
    // A 16x32: lane<16 holds K0-7,K16-23 of row l; lane>=16 holds K8-15,K24-31
    v16bf af[4];
#pragma unroll
    for (int mt = 0; mt < 4; ++mt) {
      const __bf16* p = lsA[cur] + (waveM * 64 + mt * 16 + l) * LDT;
      union { v16bf w; v8bf h[2]; } u;
      u.h[0] = *(const v8bf*)(p + half * 8);
      u.h[1] = *(const v8bf*)(p + 16 + half * 8);
      af[mt] = u.w;
    }
    // B 32x16: lane<16 holds K0-7,K16-23 of col l; lane>=16 holds K8-15,K24-31
    v16bf bfr[4];
#pragma unroll
    for (int nt = 0; nt < 4; ++nt) {
      const __bf16* p = lsB[cur] + (waveN * 64 + nt * 16 + l) * LDT;
      union { v16bf w; v8bf h[2]; } u;
      u.h[0] = *(const v8bf*)(p + half * 8);
      u.h[1] = *(const v8bf*)(p + 16 + half * 8);
      bfr[nt] = u.w;
    }

#pragma unroll
    for (int mt = 0; mt < 4; ++mt)
#pragma unroll
      for (int nt = 0; nt < 4; ++nt)
        acc[mt][nt] = __builtin_amdgcn_wmma_f32_16x16x32_bf16(
            false, af[mt], false, bfr[nt], (short)0, acc[mt][nt], false, false);
  }

  // ---- epilogue: D layout: VGPR g -> (M = g + 8*half, N = l) in 16x16 tile
#pragma unroll
  for (int mt = 0; mt < 4; ++mt) {
#pragma unroll
    for (int nt = 0; nt < 4; ++nt) {
      int col = blockN + waveN * 64 + nt * 16 + l;
      float bv = HAS_BIAS ? bias[col] : 0.0f;
#pragma unroll
      for (int g = 0; g < 8; ++g) {
        int row = blockM + waveM * 64 + mt * 16 + half * 8 + g;
        float v = acc[mt][nt][g] + bv;
        if (RELU) v = fmaxf(v, 0.0f);
        if (WF) outF[(size_t)row * N + col] = v;
        if (WBMODE == 1) outB[(size_t)row * N + col] = (__bf16)v;
        if (WBMODE == 2) outB[(size_t)col * M + row] = (__bf16)v;  // B^T feed
      }
    }
  }
}

// ---------------------------------------------------------------------------
// Routing head: one wave32 per node.
//   scores s[e] = hc . Wp[:,e]  (hc = [h2relu | hdst], 1024 dims)
//   idx = argmax_e s[e];  out[c] = hc . Wpp[idx][:, c]   (Wpp: 0.9MB -> L2)
// ---------------------------------------------------------------------------
__global__ __launch_bounds__(256)
void gppt_head(const float* __restrict__ h2,   // relu(layer1 out) [Nn,512]
               const float* __restrict__ hd,   // h_dst (already relu) [Nn,512]
               const float* __restrict__ Wp,   // [1024,7]
               const float* __restrict__ Wpp,  // [7,1024,32]
               float* __restrict__ out)        // [Nn,32]
{
  const int lane = threadIdx.x & 31;
  const int wave = threadIdx.x >> 5;
  const int node = blockIdx.x * 8 + wave;

  const float* a = h2 + (size_t)node * 512;
  const float* b = hd + (size_t)node * 512;

  float s[7] = {0.f, 0.f, 0.f, 0.f, 0.f, 0.f, 0.f};
  for (int d = lane; d < 512; d += 32) {
    float va = a[d], vb = b[d];
    const float* w0 = Wp + (size_t)d * 7;
    const float* w1 = Wp + (size_t)(d + 512) * 7;
#pragma unroll
    for (int e = 0; e < 7; ++e) s[e] += va * w0[e] + vb * w1[e];
  }
#pragma unroll
  for (int e = 0; e < 7; ++e)
#pragma unroll
    for (int off = 16; off > 0; off >>= 1)
      s[e] += __shfl_down(s[e], off, 32);

  int idx = 0;
  if (lane == 0) {
    float best = s[0];
#pragma unroll
    for (int e = 1; e < 7; ++e)
      if (s[e] > best) { best = s[e]; idx = e; }   // first-max, matches argmax
  }
  idx = __shfl(idx, 0, 32);

  const float* w = Wpp + (size_t)idx * 1024 * 32 + lane;  // lane = out channel
  float acc = 0.f;
  for (int d = 0; d < 512; ++d) {
    acc += a[d] * w[(size_t)d * 32];
    acc += b[d] * w[(size_t)(d + 512) * 32];
  }
  out[(size_t)node * 32 + lane] = acc;
}

// ---------------------------------------------------------------------------
// Orchestration
// ---------------------------------------------------------------------------
extern "C" void kernel_launch(void* const* d_in, const int* in_sizes, int n_in,
                              void* d_out, int out_size, void* d_ws, size_t ws_size,
                              hipStream_t stream) {
  const float* feature = (const float*)d_in[0]; // [8192,512]
  const float* adj     = (const float*)d_in[1]; // [8192,8192]
  const float* W0      = (const float*)d_in[2]; // [512,512]
  const float* b0      = (const float*)d_in[3]; // [512]
  const float* W1      = (const float*)d_in[4]; // [512,512]
  const float* b1      = (const float*)d_in[5]; // [512]
  const float* Wp      = (const float*)d_in[6]; // [1024,7]
  const float* Wpp     = (const float*)d_in[7]; // [7,1024,32]
  float* out = (float*)d_out;                   // [8192,32]

  const long long Nn = 8192, Hd = 512;

  // workspace layout (bytes)
  char* ws = (char*)d_ws;
  __bf16* adjB = (__bf16*)(ws);                                    // 128 MB
  __bf16* xBt  = (__bf16*)(ws + 134217728ull);                     //   8 MB (t0^T/t1^T [512,8192])
  float*  hF   = (float*) (ws + 134217728ull + 8388608ull);        //  16 MB (h = h_dst)
  float*  h2F  = (float*) (ws + 134217728ull + 8388608ull + 16777216ull); // 16 MB
  __bf16* fB   = (__bf16*)(ws + 134217728ull + 8388608ull + 33554432ull); //  8 MB (feat/h bf16)
  __bf16* wBt  = (__bf16*)(ws + 134217728ull + 8388608ull + 33554432ull + 8388608ull); // 0.5 MB (W^T)

  dim3 blk(256);
  dim3 gGemm(8192 / BM, 512 / BN);   // 32 x 4 workgroups

  // adj -> bf16 (128 MB: fits 192MB L2 for the 2nd big GEMM pass)
  gppt_cvt_f32_bf16<<<(Nn * Nn) / (8 * 256), blk, 0, stream>>>(adj, adjB, Nn * Nn);
  gppt_cvt_f32_bf16<<<(Nn * Hd) / (8 * 256), blk, 0, stream>>>(feature, fB, Nn * Hd);
  gppt_cvt_f32_bf16_T<<<(Hd * Hd) / 256, blk, 0, stream>>>(W0, wBt, Hd, Hd);

  // t0^T = (feature @ W0)^T              [512,8192] bf16
  gppt_gemm_bf16<false,false,false,2><<<gGemm, blk, 0, stream>>>(
      fB, wBt, nullptr, nullptr, xBt, 8192, 512, 512);
  // h = relu(adj @ t0 + b0)  -> f32 (h_dst) + bf16 row-major (A of next GEMM)
  gppt_gemm_bf16<true,true,true,1><<<gGemm, blk, 0, stream>>>(
      adjB, xBt, b0, hF, fB, 8192, 512, 8192);
  // W1^T (reuse wBt)
  gppt_cvt_f32_bf16_T<<<(Hd * Hd) / 256, blk, 0, stream>>>(W1, wBt, Hd, Hd);
  // t1^T = (h @ W1)^T                    [512,8192] bf16 (reuse xBt)
  gppt_gemm_bf16<false,false,false,2><<<gGemm, blk, 0, stream>>>(
      fB, wBt, nullptr, nullptr, xBt, 8192, 512, 512);
  // h2 = relu(adj @ t1 + b1) -> f32
  gppt_gemm_bf16<true,true,true,0><<<gGemm, blk, 0, stream>>>(
      adjB, xBt, b1, h2F, nullptr, 8192, 512, 8192);
  // routing head (relu(h_dst) is identity: hF already relu'd)
  gppt_head<<<8192 / 8, blk, 0, stream>>>(h2F, hF, Wp, Wpp, out);
}
